// GraphMHC_54528904790553
// MI455X (gfx1250) — matrix operand order repaired and checked by reading, not compiled, not run on hardware
//
#include <hip/hip_runtime.h>
#include <hip/hip_bf16.h>

// ---------------- problem constants ----------------
constexpr int N_  = 20000;
constexpr int E_  = 320000;
constexpr int EF_ = E_ + N_;      // with self loops
constexpr int B_  = 64;
constexpr int CH_ = 64;
constexpr int Hh_ = 4;
constexpr int HC_ = 256;          // H*CH
constexpr int ED_ = 16;

typedef __attribute__((ext_vector_type(2)))  float    v2f;
typedef __attribute__((ext_vector_type(8)))  float    v8f;
typedef __attribute__((ext_vector_type(16))) _Float16 v16h;

#if defined(__has_builtin)
#  if __has_builtin(__builtin_amdgcn_wmma_f32_16x16x4_f32)
#    define USE_WMMA_F32 1
#  endif
#endif
#ifndef USE_WMMA_F32
#  define USE_WMMA_F32 0
#endif

#define GID() (blockIdx.x * blockDim.x + threadIdx.x)
static inline unsigned ceil_div(long a, long b) { return (unsigned)((a + b - 1) / b); }

// sign-aware float atomic max (init must be -inf)
__device__ __forceinline__ void atomicMaxF(float* addr, float val) {
    if (val >= 0.0f) atomicMax((int*)addr, __float_as_int(val));
    else             atomicMin((unsigned int*)addr, __float_as_uint(val));
}

// ---------------- utility kernels ----------------
__global__ void k_zero(float* p, int n) { int i = GID(); if (i < n) p[i] = 0.0f; }
__global__ void k_fill(float* p, int n, float v) { int i = GID(); if (i < n) p[i] = v; }
__global__ void k_copy_int(const int* __restrict__ s, int* __restrict__ d, int n) {
    int i = GID(); if (i < n) d[i] = s[i];
}

// ---------------- self-loop attr: deg + per-dst mean of incoming edge_attr ----------------
__global__ void k_loop_acc(const float* __restrict__ ea, const int* __restrict__ col,
                           float* __restrict__ deg, float* __restrict__ loopattr) {
    int i = GID();
    if (i >= E_ * ED_) return;
    int e = i >> 4, d = i & 15;
    int c = col[e];
    atomicAdd(&loopattr[c * ED_ + d], ea[i]);
    if (d == 0) atomicAdd(&deg[c], 1.0f);
}
__global__ void k_loop_fin(float* __restrict__ loopattr, const float* __restrict__ deg) {
    int i = GID();
    if (i >= N_ * ED_) return;
    loopattr[i] /= fmaxf(deg[i >> 4], 1.0f);
}

// ---------------- CSR (by destination) build: done ONCE, reused by all 4 layers ----------------
__global__ void k_cnt_from_deg(const float* __restrict__ deg, int* __restrict__ cnt) {
    int i = GID();
    if (i >= N_) return;
    cnt[i] = (int)(deg[i] + 0.5f) + 1;      // + self loop
}
// single block 1024-thread exclusive scan -> rowptr[N_+1]
__global__ __launch_bounds__(1024) void k_scan_rowptr(const int* __restrict__ cnt,
                                                      int* __restrict__ rowptr) {
    __shared__ int part[1024];
    int tid = threadIdx.x;
    const int CHUNK = (N_ + 1023) / 1024;   // 20
    int base = tid * CHUNK;
    int s = 0;
    for (int j = 0; j < CHUNK; ++j) { int n = base + j; if (n < N_) s += cnt[n]; }
    part[tid] = s;
    __syncthreads();
    for (int off = 1; off < 1024; off <<= 1) {
        int add = (tid >= off) ? part[tid - off] : 0;
        __syncthreads();
        part[tid] += add;
        __syncthreads();
    }
    int run = (tid == 0) ? 0 : part[tid - 1];
    for (int j = 0; j < CHUNK; ++j) {
        int n = base + j;
        if (n < N_) { rowptr[n] = run; run += cnt[n]; }
    }
    if (tid == 1023) rowptr[N_] = run;      // == EF_
}
__global__ void k_csr_fill(const int* __restrict__ col, int* __restrict__ cursor,
                           int* __restrict__ eids) {
    int i = GID();
    if (i >= EF_) return;
    int dst = (i < E_) ? col[i] : i - E_;
    int pos = atomicAdd(&cursor[dst], 1);
    eids[pos] = i;
}

// ---------------- WMMA GEMM: H[N,256] = X[N,64] @ W[64,256] ----------------
// One wave computes a 16x64 strip (4 tiles), reusing the A fragment 4x per K-step.
__global__ __launch_bounds__(256) void k_gemm_xw(const float* __restrict__ X,
                                                 const float* __restrict__ W,
                                                 float* __restrict__ Hout) {
    int wave = blockIdx.x * (blockDim.x >> 5) + (threadIdx.x >> 5);
    int tm = wave >> 2;          // 1250 row tiles
    int tq = wave & 3;           // 64-col quadrant
    if (tm >= N_ / 16) return;   // wave-uniform
    int lane = threadIdx.x & 31;
    int half = lane >> 4;
    int l16  = lane & 15;
    v8f acc0 = {}, acc1 = {}, acc2 = {}, acc3 = {};
    const float* Xr = X + (size_t)(tm * 16 + l16) * CH_;
    const float* Wc = W + tq * 64 + l16;
#if USE_WMMA_F32
    #pragma unroll
    for (int k = 0; k < CH_; k += 4) {
        v2f a, b0, b1, b2, b3;
        a.x = Xr[k + 2 * half];
        a.y = Xr[k + 2 * half + 1];
        const float* w0 = Wc + (size_t)(k + 2 * half) * HC_;
        const float* w1 = Wc + (size_t)(k + 2 * half + 1) * HC_;
        b0.x = w0[0];  b0.y = w1[0];
        b1.x = w0[16]; b1.y = w1[16];
        b2.x = w0[32]; b2.y = w1[32];
        b3.x = w0[48]; b3.y = w1[48];
        acc0 = __builtin_amdgcn_wmma_f32_16x16x4_f32(false, a, false, b0, (short)0, acc0, false, false);
        acc1 = __builtin_amdgcn_wmma_f32_16x16x4_f32(false, a, false, b1, (short)0, acc1, false, false);
        acc2 = __builtin_amdgcn_wmma_f32_16x16x4_f32(false, a, false, b2, (short)0, acc2, false, false);
        acc3 = __builtin_amdgcn_wmma_f32_16x16x4_f32(false, a, false, b3, (short)0, acc3, false, false);
    }
#else
    #pragma unroll
    for (int kc = 0; kc < CH_; kc += 32) {
        v16h a16, b0, b1, b2, b3;
        #pragma unroll
        for (int v = 0; v < 8; ++v) {
            int ka = kc + (v < 4 ? 2 * v : 16 + 2 * (v - 4)) + 8 * half;
            a16[2 * v]     = (_Float16)Xr[ka];
            a16[2 * v + 1] = (_Float16)Xr[ka + 1];
            int kb = kc + (v < 4 ? 2 * v : 16 + 2 * (v - 4)) + 8 * half;
            const float* w0 = Wc + (size_t)kb * HC_;
            const float* w1 = Wc + (size_t)(kb + 1) * HC_;
            b0[2 * v] = (_Float16)w0[0];  b0[2 * v + 1] = (_Float16)w1[0];
            b1[2 * v] = (_Float16)w0[16]; b1[2 * v + 1] = (_Float16)w1[16];
            b2[2 * v] = (_Float16)w0[32]; b2[2 * v + 1] = (_Float16)w1[32];
            b3[2 * v] = (_Float16)w0[48]; b3[2 * v + 1] = (_Float16)w1[48];
        }
        acc0 = __builtin_amdgcn_wmma_f32_16x16x32_f16(false, a16, false, b0, (short)0, acc0, false, false);
        acc1 = __builtin_amdgcn_wmma_f32_16x16x32_f16(false, a16, false, b1, (short)0, acc1, false, false);
        acc2 = __builtin_amdgcn_wmma_f32_16x16x32_f16(false, a16, false, b2, (short)0, acc2, false, false);
        acc3 = __builtin_amdgcn_wmma_f32_16x16x32_f16(false, a16, false, b3, (short)0, acc3, false, false);
    }
#endif
    float* Op = Hout + (size_t)(tm * 16) * HC_ + tq * 64 + l16;
    #pragma unroll
    for (int j = 0; j < 8; ++j) {
        size_t r = (size_t)(j + 8 * half) * HC_;
        Op[r + 0]  = acc0[j];
        Op[r + 16] = acc1[j];
        Op[r + 32] = acc2[j];
        Op[r + 48] = acc3[j];
    }
}

// ---------------- per-node attention projections ----------------
__global__ void k_node_att(const float* __restrict__ hbuf, const float* __restrict__ att_src,
                           const float* __restrict__ att_dst, float* __restrict__ as_,
                           float* __restrict__ ad_) {
    int i = GID();
    if (i >= N_ * Hh_) return;
    int n = i >> 2, hh = i & 3;
    const float* hp = hbuf + (size_t)n * HC_ + hh * CH_;
    const float* s = att_src + hh * CH_;
    const float* d = att_dst + hh * CH_;
    float a = 0.f, b = 0.f;
    for (int c = 0; c < CH_; ++c) { a += hp[c] * s[c]; b += hp[c] * d[c]; }
    as_[i] = a; ad_[i] = b;
}

// M[d,h] = sum_c We[d, h*64+c] * att_edge[h,c]   (collapses the ea@We GEMM)
__global__ void k_edge_M(const float* __restrict__ We, const float* __restrict__ att_edge,
                         float* __restrict__ M) {
    int i = threadIdx.x;
    if (i >= ED_ * Hh_) return;
    int d = i >> 2, hh = i & 3;
    float acc = 0.f;
    for (int c = 0; c < CH_; ++c) acc += We[d * HC_ + hh * CH_ + c] * att_edge[hh * CH_ + c];
    M[d * Hh_ + hh] = acc;
}

// ---------------- edge logits (coalesced edge-centric, no atomics) ----------------
__global__ void k_edge_logit(const float* __restrict__ ea, const float* __restrict__ loopattr,
                             const int* __restrict__ row, const int* __restrict__ col,
                             const float* __restrict__ as_, const float* __restrict__ ad_,
                             const float* __restrict__ M, float* __restrict__ alpha) {
    int i = GID();
    if (i >= EF_) return;
    int r, c; const float* eap;
    if (i < E_) { r = row[i]; c = col[i]; eap = ea + (size_t)i * ED_; }
    else        { r = c = i - E_;         eap = loopattr + (size_t)(i - E_) * ED_; }
    float aeh[4] = {0.f, 0.f, 0.f, 0.f};
    #pragma unroll
    for (int d = 0; d < ED_; ++d) {
        float v = eap[d];
        #pragma unroll
        for (int hh = 0; hh < 4; ++hh) aeh[hh] += v * M[d * 4 + hh];
    }
    #pragma unroll
    for (int hh = 0; hh < 4; ++hh) {
        float a = as_[r * 4 + hh] + ad_[c * 4 + hh] + aeh[hh];
        alpha[(size_t)i * 4 + hh] = a > 0.f ? a : 0.2f * a;   // leaky relu
    }
}

// ---------------- node-centric segmented softmax over CSR (no atomics) ----------------
__global__ void k_node_softmax(const int* __restrict__ rowptr, const int* __restrict__ eids,
                               float* __restrict__ alpha, float* __restrict__ sinv) {
    int i = GID();
    if (i >= N_ * Hh_) return;
    int n = i >> 2, hh = i & 3;
    int jb = rowptr[n], je = rowptr[n + 1];
    float m = -3.4e38f;
    for (int j = jb; j < je; ++j) m = fmaxf(m, alpha[(size_t)eids[j] * 4 + hh]);
    float s = 0.f;
    for (int j = jb; j < je; ++j) {
        size_t idx = (size_t)eids[j] * 4 + hh;
        float a = __expf(alpha[idx] - m);
        alpha[idx] = a;
        s += a;
    }
    sinv[i] = 1.0f / (s + 1e-16f);
}

// ---------------- node-centric aggregation + head-mean + bias + GN sums (no big atomics) ----------------
__global__ void k_node_gather(const int* __restrict__ rowptr, const int* __restrict__ eids,
                              const int* __restrict__ row, const float* __restrict__ alpha,
                              const float* __restrict__ sinv, const float* __restrict__ hbuf,
                              const float* __restrict__ bias, const int* __restrict__ batch,
                              float* __restrict__ xtmp, float* __restrict__ gnsum,
                              float* __restrict__ gncnt) {
    int i = GID();
    if (i >= N_ * CH_) return;
    int n = i >> 6, c = i & 63;
    int jb = rowptr[n], je = rowptr[n + 1];
    float a0 = 0.f, a1 = 0.f, a2 = 0.f, a3 = 0.f;
    for (int j = jb; j < je; ++j) {
        int e = eids[j];
        int r = (e < E_) ? row[e] : e - E_;
        if (j + 1 < je) {                      // prefetch next gathered row
            int e2 = eids[j + 1];
            int r2 = (e2 < E_) ? row[e2] : e2 - E_;
            __builtin_prefetch(hbuf + (size_t)r2 * HC_ + c, 0, 0);
        }
        const float* hp = hbuf + (size_t)r * HC_ + c;
        const float* ap = alpha + (size_t)e * 4;
        a0 += ap[0] * hp[0];
        a1 += ap[1] * hp[64];
        a2 += ap[2] * hp[128];
        a3 += ap[3] * hp[192];
    }
    const float* sp = sinv + n * 4;
    float v = 0.25f * (a0 * sp[0] + a1 * sp[1] + a2 * sp[2] + a3 * sp[3]) + bias[c];
    xtmp[i] = v;
    int b = batch[n];
    atomicAdd(&gnsum[b * CH_ + c], v);
    if (c == 0) atomicAdd(&gncnt[b], 1.0f);
}

// ---------------- GraphNorm ----------------
__global__ void k_gn_div(float* __restrict__ p, const float* __restrict__ cnt) {
    int i = GID();
    if (i >= B_ * CH_) return;
    p[i] /= fmaxf(cnt[i >> 6], 1.0f);
}
__global__ void k_gn_center(float* __restrict__ xtmp, const float* __restrict__ gnmean,
                            const float* __restrict__ mean_scale, const int* __restrict__ batch,
                            float* __restrict__ gnvar) {
    int i = GID();
    if (i >= N_ * CH_) return;
    int n = i >> 6, c = i & 63;
    int b = batch[n];
    float xc = xtmp[i] - mean_scale[c] * gnmean[b * CH_ + c];
    xtmp[i] = xc;
    atomicAdd(&gnvar[b * CH_ + c], xc * xc);
}
__global__ void k_gn_final(const float* __restrict__ xtmp, const float* __restrict__ gnvar,
                           const float* __restrict__ w, const float* __restrict__ bb,
                           const int* __restrict__ batch, float* __restrict__ xcur) {
    int i = GID();
    if (i >= N_ * CH_) return;
    int n = i >> 6, c = i & 63;
    int b = batch[n];
    float y = w[c] * xtmp[i] * rsqrtf(gnvar[b * CH_ + c] + 1e-5f) + bb[c];
    xcur[i] = y > 0.f ? y : 0.f;
}

// ---------------- global max pool ----------------
__global__ void k_pool_max(const float* __restrict__ xcur, const int* __restrict__ batch,
                           float* __restrict__ gmax) {
    int i = GID();
    if (i >= N_ * CH_) return;
    int n = i >> 6, c = i & 63;
    atomicMaxF(&gmax[batch[n] * CH_ + c], xcur[i]);
}

// ---------------- head: conv5/bn5/pool -> conv6/bn6/pool + skips + lin7 (tiny, 1 WG) ----------------
__global__ __launch_bounds__(256) void k_head(const float* __restrict__ g,
        const float* __restrict__ w5, const float* __restrict__ b5,
        const float* __restrict__ g5, const float* __restrict__ bb5,
        const float* __restrict__ w6, const float* __restrict__ b6,
        const float* __restrict__ g6, const float* __restrict__ bb6,
        const float* __restrict__ lw, const float* __restrict__ lb,
        float* __restrict__ out) {
    __shared__ float y1[64 * 8 * 16];   // 8192 floats; reused for conv6 output [64][16][8]
    __shared__ float y2[64 * 8 * 8];    // 4096 floats (= skip5 flat [64][64])
    __shared__ float red[32];
    int tid = threadIdx.x;

    for (int idx = tid; idx < 64 * 8 * 16; idx += 256) {   // conv5 SAME, in g=[64][4][16]
        int b = idx >> 7, o = (idx >> 4) & 7, t = idx & 15;
        float acc = b5[o];
        for (int ic = 0; ic < 4; ++ic)
            for (int k = 0; k < 3; ++k) {
                int tt = t + k - 1;
                if (tt >= 0 && tt < 16) acc += w5[(o * 4 + ic) * 3 + k] * g[b * 64 + ic * 16 + tt];
            }
        y1[idx] = acc;
    }
    if (tid < 32) red[tid] = 0.0f;
    __syncthreads();
    for (int idx = tid; idx < 8192; idx += 256) {          // bn5 stats (1024/ch)
        int o = (idx >> 4) & 7;
        float v = y1[idx];
        atomicAdd(&red[o], v);
        atomicAdd(&red[16 + o], v * v);
    }
    __syncthreads();
    for (int idx = tid; idx < 64 * 8 * 8; idx += 256) {    // bn5+relu+avgpool -> y2
        int b = idx >> 6, o = (idx >> 3) & 7, t = idx & 7;
        float mu = red[o] * (1.0f / 1024.0f);
        float var = red[16 + o] * (1.0f / 1024.0f) - mu * mu;
        float is = rsqrtf(fmaxf(var, 0.0f) + 1e-5f);
        float v0 = g5[o] * (y1[(b * 8 + o) * 16 + 2 * t] - mu) * is + bb5[o];
        float v1 = g5[o] * (y1[(b * 8 + o) * 16 + 2 * t + 1] - mu) * is + bb5[o];
        v0 = v0 > 0.f ? v0 : 0.f; v1 = v1 > 0.f ? v1 : 0.f;
        y2[idx] = 0.5f * (v0 + v1);
    }
    __syncthreads();
    for (int idx = tid; idx < 64 * 16 * 8; idx += 256) {   // conv6 -> y1 reused [64][16][8]
        int b = idx >> 7, o = (idx >> 3) & 15, t = idx & 7;
        float acc = b6[o];
        for (int ic = 0; ic < 8; ++ic)
            for (int k = 0; k < 3; ++k) {
                int tt = t + k - 1;
                if (tt >= 0 && tt < 8) acc += w6[(o * 8 + ic) * 3 + k] * y2[(b * 8 + ic) * 8 + tt];
            }
        y1[idx] = acc;
    }
    __syncthreads();
    if (tid < 32) red[tid] = 0.0f;
    __syncthreads();
    for (int idx = tid; idx < 8192; idx += 256) {          // bn6 stats (512/ch)
        int o = (idx >> 3) & 15;
        float v = y1[idx];
        atomicAdd(&red[o], v);
        atomicAdd(&red[16 + o], v * v);
    }
    __syncthreads();
    if (tid < 64) {                                        // bn6+relu+pool+skips+lin7
        int b = tid;
        float acc = lb[0];
        for (int o = 0; o < 16; ++o) {
            float mu = red[o] * (1.0f / 512.0f);
            float var = red[16 + o] * (1.0f / 512.0f) - mu * mu;
            float is = rsqrtf(fmaxf(var, 0.0f) + 1e-5f);
            for (int t = 0; t < 4; ++t) {
                float v0 = g6[o] * (y1[(b * 16 + o) * 8 + 2 * t] - mu) * is + bb6[o];
                float v1 = g6[o] * (y1[(b * 16 + o) * 8 + 2 * t + 1] - mu) * is + bb6[o];
                v0 = v0 > 0.f ? v0 : 0.f; v1 = v1 > 0.f ? v1 : 0.f;
                int c = o * 4 + t;
                float val = 0.5f * (v0 + v1) + g[b * 64 + c] + y2[b * 64 + c];
                acc += val * lw[c];
            }
        }
        out[b] = acc;
    }
}

// ---------------- host launcher ----------------
extern "C" void kernel_launch(void* const* d_in, const int* in_sizes, int n_in,
                              void* d_out, int out_size, void* d_ws, size_t ws_size,
                              hipStream_t stream) {
    (void)in_sizes; (void)n_in; (void)out_size; (void)ws_size;
    const float* x     = (const float*)d_in[0];
    const float* ea    = (const float*)d_in[1];
    const int*   eidx  = (const int*)d_in[2];
    const int*   batch = (const int*)d_in[4];
    const int* row = eidx;
    const int* col = eidx + E_;

    // workspace carve-up
    float* ws = (float*)d_ws;
    size_t off = 0;
    float* deg      = ws + off; off += N_;
    float* loopattr = ws + off; off += (size_t)N_ * ED_;
    float* hbuf     = ws + off; off += (size_t)N_ * HC_;
    float* xcur     = ws + off; off += (size_t)N_ * CH_;
    float* xtmp     = ws + off; off += (size_t)N_ * CH_;
    float* as_      = ws + off; off += (size_t)N_ * Hh_;
    float* ad_      = ws + off; off += (size_t)N_ * Hh_;
    float* alpha    = ws + off; off += (size_t)EF_ * Hh_;
    float* sinv     = ws + off; off += (size_t)N_ * Hh_;
    float* Msmall   = ws + off; off += ED_ * Hh_;
    float* gnsum    = ws + off; off += B_ * CH_;
    float* gnvar    = ws + off; off += B_ * CH_;
    float* gncnt    = ws + off; off += B_;
    float* gmax     = ws + off; off += B_ * CH_;
    int* iws    = (int*)(ws + off);
    size_t io = 0;
    int* cnt    = iws + io; io += N_;
    int* rowptr = iws + io; io += N_ + 1;
    int* cursor = iws + io; io += N_;
    int* eids   = iws + io; io += EF_;

    const int T = 256;
    const float NEG = -3.4e38f;

    // ---- self-loop attrs + CSR build (once; col is layer-invariant) ----
    k_zero<<<ceil_div(N_ * (1 + ED_), T), T, 0, stream>>>(deg, N_ * (1 + ED_));
    k_loop_acc<<<ceil_div((long)E_ * ED_, T), T, 0, stream>>>(ea, col, deg, loopattr);
    k_loop_fin<<<ceil_div((long)N_ * ED_, T), T, 0, stream>>>(loopattr, deg);
    k_cnt_from_deg<<<ceil_div(N_, T), T, 0, stream>>>(deg, cnt);
    k_scan_rowptr<<<1, 1024, 0, stream>>>(cnt, rowptr);
    k_copy_int<<<ceil_div(N_, T), T, 0, stream>>>(rowptr, cursor, N_);
    k_csr_fill<<<ceil_div(EF_, T), T, 0, stream>>>(col, cursor, eids);

    for (int l = 0; l < 4; ++l) {
        const float* Wl    = (const float*)d_in[5 + l * 9 + 0];
        const float* Wel   = (const float*)d_in[5 + l * 9 + 1];
        const float* asrc  = (const float*)d_in[5 + l * 9 + 2];
        const float* adst  = (const float*)d_in[5 + l * 9 + 3];
        const float* aedg  = (const float*)d_in[5 + l * 9 + 4];
        const float* gbias = (const float*)d_in[5 + l * 9 + 5];
        const float* nw    = (const float*)d_in[5 + l * 9 + 6];
        const float* nb    = (const float*)d_in[5 + l * 9 + 7];
        const float* nms   = (const float*)d_in[5 + l * 9 + 8];
        const float* xin = (l == 0) ? x : xcur;

        // h = x @ W (WMMA): 1250 row tiles * 4 quadrant waves = 5000 waves
        k_gemm_xw<<<625, 256, 0, stream>>>(xin, Wl, hbuf);
        k_node_att<<<ceil_div((long)N_ * Hh_, T), T, 0, stream>>>(hbuf, asrc, adst, as_, ad_);
        k_edge_M<<<1, 64, 0, stream>>>(Wel, aedg, Msmall);
        k_zero<<<ceil_div((long)(2 * B_ * CH_ + B_), T), T, 0, stream>>>(gnsum, 2 * B_ * CH_ + B_);

        k_edge_logit<<<ceil_div(EF_, T), T, 0, stream>>>(ea, loopattr, row, col, as_, ad_,
                                                         Msmall, alpha);
        k_node_softmax<<<ceil_div((long)N_ * Hh_, T), T, 0, stream>>>(rowptr, eids, alpha, sinv);
        k_node_gather<<<ceil_div((long)N_ * CH_, T), T, 0, stream>>>(rowptr, eids, row, alpha,
                                                                     sinv, hbuf, gbias, batch,
                                                                     xtmp, gnsum, gncnt);

        k_gn_div<<<ceil_div(B_ * CH_, T), T, 0, stream>>>(gnsum, gncnt);
        k_gn_center<<<ceil_div((long)N_ * CH_, T), T, 0, stream>>>(xtmp, gnsum, nms, batch, gnvar);
        k_gn_div<<<ceil_div(B_ * CH_, T), T, 0, stream>>>(gnvar, gncnt);
        k_gn_final<<<ceil_div((long)N_ * CH_, T), T, 0, stream>>>(xtmp, gnvar, nw, nb, batch, xcur);
    }

    // global max pool + head
    k_fill<<<ceil_div(B_ * CH_, T), T, 0, stream>>>(gmax, B_ * CH_, NEG);
    k_pool_max<<<ceil_div((long)N_ * CH_, T), T, 0, stream>>>(xcur, batch, gmax);
    k_head<<<1, 256, 0, stream>>>(gmax,
                                  (const float*)d_in[41], (const float*)d_in[42],
                                  (const float*)d_in[43], (const float*)d_in[44],
                                  (const float*)d_in[45], (const float*)d_in[46],
                                  (const float*)d_in[47], (const float*)d_in[48],
                                  (const float*)d_in[49], (const float*)d_in[50],
                                  (float*)d_out);
}
